// SRU_Layer_90855738180004
// MI455X (gfx1250) — matrix operand order repaired and checked by reading, not compiled, not run on hardware
//
#include <hip/hip_runtime.h>
#include <hip/hip_bf16.h>

// ---------------------------------------------------------------------------
// SRU layer for MI455X (gfx1250):
//   U = x @ W  (f16 WMMA, fp32 accum, async-to-LDS double-buffered GEMM)
//   -> sequential SRU scan over S.
// Dims from reference: S=2048, B=16, D=H=1024.
// ---------------------------------------------------------------------------

typedef __attribute__((ext_vector_type(4)))  _Float16 v4h;
typedef __attribute__((ext_vector_type(8)))  _Float16 v8h;
typedef __attribute__((ext_vector_type(16))) _Float16 v16h;
typedef __attribute__((ext_vector_type(8)))  float    v8f;
typedef __attribute__((ext_vector_type(4)))  int      i32x4;
typedef __attribute__((address_space(1)))    i32x4*   i32x4_g;
typedef __attribute__((address_space(3)))    i32x4*   i32x4_s;

constexpr int S = 2048, Bb = 16, D = 1024, H = 1024;
constexpr int M = S * Bb;      // 32768 rows of GEMM
constexpr int N = 3 * H;       // 3072 cols
constexpr int K = D;           // 1024

constexpr int BM = 128, BN = 128, BK = 64;
constexpr int LDS_LD = BK + 8; // 72 halves = 144B row stride: 16B aligned, dodges bank conflicts

// CDNA5 async global->LDS path (ASYNCcnt), if this toolchain exposes it.
#if __has_builtin(__builtin_amdgcn_global_load_async_to_lds_b128)
#define HAVE_ASYNC_LDS 1
#endif

__device__ __forceinline__ void wait_async0() {
#if __has_builtin(__builtin_amdgcn_s_wait_asynccnt)
    __builtin_amdgcn_s_wait_asynccnt(0);
#else
    asm volatile("s_wait_asynccnt 0x0" ::: "memory");
#endif
}

// ---- kernel 1: x (fp32) -> f16 ---------------------------------------------
__global__ void cvt_x_f16(const float* __restrict__ x, _Float16* __restrict__ xh) {
    size_t i = ((size_t)blockIdx.x * blockDim.x + threadIdx.x) * 4;
    float4 v = *(const float4*)(x + i);
    v4h o = { (_Float16)v.x, (_Float16)v.y, (_Float16)v.z, (_Float16)v.w };
    *(v4h*)(xh + i) = o;
}

// ---- kernel 2: W (D x 3H, fp32) -> W^T (3H x D, f16) -----------------------
__global__ void cvt_wT_f16(const float* __restrict__ w, _Float16* __restrict__ wt) {
    __shared__ float tile[32][33];
    int n0 = blockIdx.x * 32, k0 = blockIdx.y * 32;
    #pragma unroll
    for (int i = 0; i < 4; ++i) {
        int k = threadIdx.y + i * 8;
        tile[k][threadIdx.x] = w[(size_t)(k0 + k) * N + n0 + threadIdx.x];
    }
    __syncthreads();
    #pragma unroll
    for (int i = 0; i < 4; ++i) {
        int n = threadIdx.y + i * 8;
        wt[(size_t)(n0 + n) * K + k0 + threadIdx.x] = (_Float16)tile[threadIdx.x][n];
    }
}

// ---- kernel 3: tiled WMMA GEMM  C[M][N] = A[M][K] * Bt[N][K]^T -------------
// 256 threads = 8 wave32s in a 2(M) x 4(N) grid; each wave: 64x32 = 4x2 WMMA
// tiles. Double-buffered LDS; staging via GLOBAL_LOAD_ASYNC_TO_LDS_B128 when
// available (one s_wait_asynccnt + one barrier per K-stage).
__global__ __launch_bounds__(256) void gemm_wmma_f16(
        const _Float16* __restrict__ A, const _Float16* __restrict__ Bt,
        float* __restrict__ C) {
    __shared__ _Float16 sA[2][BM * LDS_LD];   // 2 x 18 KB
    __shared__ _Float16 sB[2][BN * LDS_LD];   // 2 x 18 KB

    const int tid  = threadIdx.x;
    const int lane = tid & 31;
    const int wave = tid >> 5;
    const int wm = wave >> 2;              // 0..1
    const int wn = wave & 3;               // 0..3
    const int m0 = blockIdx.y * BM;
    const int n0 = blockIdx.x * BN;

    const int lm  = lane & 15;
    // ISA 7.12.2, 16-bit A 16x32: lanes 0-15 hold K {0..7,16..23}, lanes 16-31 {8..15,24..31}
    const int hiA = (lane < 16) ? 0 : 8;
    // 16-bit B 32x16: same K split as A: lanes 0-15 hold K {0..7,16..23}, lanes 16-31 {8..15,24..31}
    const int hiB = (lane < 16) ? 0 : 8;

    // staging chunks: 128 rows x 64 halves = 1024 x 16B chunks, 4 per thread per tile
    int rowc[4], colc[4];
    #pragma unroll
    for (int p = 0; p < 4; ++p) {
        int c = tid + p * 256;
        rowc[p] = c >> 3;
        colc[p] = (c & 7) * 8;
    }

#ifdef HAVE_ASYNC_LDS
    auto stage = [&](int buf, int kt) {
        #pragma unroll
        for (int p = 0; p < 4; ++p) {
            __builtin_amdgcn_global_load_async_to_lds_b128(
                (i32x4_g)(uintptr_t)(A + (size_t)(m0 + rowc[p]) * K + kt + colc[p]),
                (i32x4_s)(uintptr_t)(&sA[buf][rowc[p] * LDS_LD + colc[p]]),
                0, 0);
            __builtin_amdgcn_global_load_async_to_lds_b128(
                (i32x4_g)(uintptr_t)(Bt + (size_t)(n0 + rowc[p]) * K + kt + colc[p]),
                (i32x4_s)(uintptr_t)(&sB[buf][rowc[p] * LDS_LD + colc[p]]),
                0, 0);
        }
    };
#else
    uint4 pfA[4], pfB[4];
    auto fetch = [&](int kt) {
        #pragma unroll
        for (int p = 0; p < 4; ++p) {
            pfA[p] = *(const uint4*)(A  + (size_t)(m0 + rowc[p]) * K + kt + colc[p]);
            pfB[p] = *(const uint4*)(Bt + (size_t)(n0 + rowc[p]) * K + kt + colc[p]);
        }
    };
    auto commit = [&](int buf) {
        #pragma unroll
        for (int p = 0; p < 4; ++p) {
            *(uint4*)(&sA[buf][rowc[p] * LDS_LD + colc[p]]) = pfA[p];
            *(uint4*)(&sB[buf][rowc[p] * LDS_LD + colc[p]]) = pfB[p];
        }
    };
#endif

    v8f acc[4][2] = {};

    auto compute = [&](int buf) {
        #pragma unroll
        for (int ks = 0; ks < BK; ks += 32) {
            v16h af[4];
            #pragma unroll
            for (int im = 0; im < 4; ++im) {
                const _Float16* p = &sA[buf][(wm * 64 + im * 16 + lm) * LDS_LD + ks + hiA];
                v8h lo = *(const v8h*)p;
                v8h hi = *(const v8h*)(p + 16);
                af[im] = __builtin_shufflevector(lo, hi, 0,1,2,3,4,5,6,7,8,9,10,11,12,13,14,15);
            }
            v16h bfr[2];
            #pragma unroll
            for (int jn = 0; jn < 2; ++jn) {
                const _Float16* p = &sB[buf][(wn * 32 + jn * 16 + lm) * LDS_LD + ks + hiB];
                v8h lo = *(const v8h*)p;
                v8h hi = *(const v8h*)(p + 16);
                bfr[jn] = __builtin_shufflevector(lo, hi, 0,1,2,3,4,5,6,7,8,9,10,11,12,13,14,15);
            }
            #pragma unroll
            for (int im = 0; im < 4; ++im)
                #pragma unroll
                for (int jn = 0; jn < 2; ++jn)
                    acc[im][jn] = __builtin_amdgcn_wmma_f32_16x16x32_f16(
                        false, af[im], false, bfr[jn], (short)0, acc[im][jn], false, false);
        }
    };

    // ---- pipelined main loop: one barrier (+async wait) per stage ----
#ifdef HAVE_ASYNC_LDS
    stage(0, 0);
    int buf = 0;
    for (int kt = 0; kt < K; kt += BK, buf ^= 1) {
        wait_async0();          // my async writes into sX[buf] have landed
        __syncthreads();        // everyone's writes visible; prev compute on buf done
        if (kt + BK < K) stage(buf ^ 1, kt + BK);
        compute(buf);
    }
#else
    fetch(0);
    commit(0);
    int buf = 0;
    for (int kt = 0; kt < K; kt += BK, buf ^= 1) {
        __syncthreads();        // commits into sX[buf] visible; prev compute on buf done
        if (kt + BK < K) fetch(kt + BK);
        compute(buf);
        if (kt + BK < K) commit(buf ^ 1);
    }
#endif

    // epilogue: 16x16 f32 C tile layout: VGPR v -> row v (lanes 0-15) / row v+8 (lanes 16-31)
    const int rlo = (lane < 16) ? 0 : 8;
    #pragma unroll
    for (int im = 0; im < 4; ++im) {
        #pragma unroll
        for (int jn = 0; jn < 2; ++jn) {
            int r0 = m0 + wm * 64 + im * 16 + rlo;
            int c0 = n0 + wn * 32 + jn * 16 + lm;
            #pragma unroll
            for (int v = 0; v < 8; ++v)
                __builtin_nontemporal_store(acc[im][jn][v], C + (size_t)(r0 + v) * N + c0);
        }
    }
}

// ---- kernel 4: SRU recurrence + highway ------------------------------------
// one thread per (b,h) chain; U layout [s*B+b][3H], column 3h+t picks gate t.
__global__ __launch_bounds__(256) void sru_scan(
        const float* __restrict__ U, const float* __restrict__ x,
        const float* __restrict__ bias, float* __restrict__ h_out,
        float* __restrict__ c_out) {
    int t = blockIdx.x * blockDim.x + threadIdx.x;   // 0 .. B*H-1
    int b = t >> 10;
    int h = t & (H - 1);
    const float fb = bias[h];
    const float rb = bias[H + h];
    float c = 0.0f;
    for (int s = 0; s < S; ++s) {
        size_t row = (size_t)(s * Bb + b);
        const float* up = U + row * N + 3 * h;
        float u0 = __builtin_nontemporal_load(up + 0);
        float u1 = __builtin_nontemporal_load(up + 1);
        float u2 = __builtin_nontemporal_load(up + 2);
        float xv = x[row * H + h];
        float f = 1.0f / (1.0f + __expf(-(u1 + fb)));
        float r = 1.0f / (1.0f + __expf(-(u2 + rb)));
        c = f * (c - u0) + u0;                               // f*c + (1-f)*u0
        float th = 2.0f / (1.0f + __expf(-2.0f * c)) - 1.0f; // tanh(c)
        __builtin_nontemporal_store(r * (th - xv) + xv, h_out + row * H + h);
    }
    c_out[(size_t)b * H + h] = c;
}

// ---------------------------------------------------------------------------
extern "C" void kernel_launch(void* const* d_in, const int* in_sizes, int n_in,
                              void* d_out, int out_size, void* d_ws, size_t ws_size,
                              hipStream_t stream) {
    const float* x    = (const float*)d_in[0];   // (S,B,D) fp32
    const float* w    = (const float*)d_in[1];   // (D,3H) fp32
    const float* bias = (const float*)d_in[2];   // (2H,)  fp32

    // workspace: Xh (64MB) | Wt (6MB) | U (384MB)
    _Float16* Xh = (_Float16*)d_ws;
    _Float16* Wt = (_Float16*)((char*)d_ws + (size_t)M * K * sizeof(_Float16));
    float*    U  = (float*)((char*)d_ws + (size_t)M * K * sizeof(_Float16)
                                        + (size_t)N * K * sizeof(_Float16));

    float* h_out = (float*)d_out;                 // (S,B,H)
    float* c_out = h_out + (size_t)S * Bb * H;    // (B,H)

    cvt_x_f16 <<< (M * K / 4) / 256, 256, 0, stream >>> (x, Xh);
    cvt_wT_f16<<< dim3(N / 32, K / 32), dim3(32, 8), 0, stream >>> (w, Wt);
    gemm_wmma_f16<<< dim3(N / BN, M / BM), 256, 0, stream >>> (Xh, Wt, U);
    sru_scan  <<< (Bb * H) / 256, 256, 0, stream >>> (U, x, bias, h_out, c_out);
}
